// PhysicsAwareAttention_54039278518829
// MI455X (gfx1250) — matrix operand order repaired and checked by reading, not compiled, not run on hardware
//
#include <hip/hip_runtime.h>

typedef __attribute__((ext_vector_type(16))) _Float16 v16h;
typedef __attribute__((ext_vector_type(8)))  float    v8f;
typedef __attribute__((ext_vector_type(4)))  unsigned int u32x4;
typedef __attribute__((ext_vector_type(8)))  int      i32x8;
typedef __attribute__((ext_vector_type(4)))  int      i32x4;

#define B_  16
#define N_  512
#define E_  128
#define H_  8
#define DH_ 16

#if __has_builtin(__builtin_amdgcn_tensor_load_to_lds) && __has_builtin(__builtin_amdgcn_s_wait_tensorcnt)
#define HAVE_TDM 1
#else
#define HAVE_TDM 0
#endif

static __device__ __forceinline__ float warpMax(float v) {
  #pragma unroll
  for (int m = 16; m >= 1; m >>= 1) v = fmaxf(v, __shfl_xor(v, m, 32));
  return v;
}
static __device__ __forceinline__ float warpSum(float v) {
  #pragma unroll
  for (int m = 16; m >= 1; m >>= 1) v += __shfl_xor(v, m, 32);
  return v;
}

// WMMA 16-bit A-operand chunks live at +0..7 and +16..23 halves from a base
// that already includes the hi-half offset: load as two b128s.
static __device__ __forceinline__ v16h load_a_tile(const _Float16* base) {
  union { uint4 u[2]; v16h v; } t;
  const uint4* p = (const uint4*)base;
  t.u[0] = p[0];   // elements 0..7
  t.u[1] = p[2];   // elements 8..15 (+16 halves)
  return t.v;
}
// 16 contiguous halves (B-operand: per-lane K-sequential run)
static __device__ __forceinline__ v16h load_row16(const _Float16* base) {
  union { uint4 u[2]; v16h v; } t;
  const uint4* p = (const uint4*)base;
  t.u[0] = p[0];
  t.u[1] = p[1];
  return t.v;
}

#if HAVE_TDM
// Tensor Data Mover: 2D tile (tile_d0 elems/row x tile_d1 rows) from global to
// LDS. Descriptor per CDNA5 ISA (groups 2/3 zero:
// tensor <= 2D). Wave-level op; EXEC ignored; tracked by TENSORcnt.
static __device__ __forceinline__ void tdm_load_2d(
    unsigned lds_byte_off, const void* gptr, unsigned data_size_log2,
    unsigned tile_d0, unsigned tile_d1, unsigned tensor_d0, unsigned tensor_d1,
    unsigned stride_d0_elems)
{
  unsigned long long ga = (unsigned long long)(uintptr_t)gptr;
  u32x4 g0;
  g0[0] = 1u;                                        // count=1 (valid), user mode
  g0[1] = lds_byte_off;                              // lds_addr
  g0[2] = (unsigned)(ga & 0xFFFFFFFFu);              // global_addr[31:0]
  g0[3] = (unsigned)((ga >> 32) & 0x01FFFFFFu) | (2u << 30);  // addr[56:32], type=2
  i32x8 g1;
  g1[0] = (int)(data_size_log2 << 16);               // wg_mask=0, data_size
  g1[1] = (int)((tensor_d0 & 0xFFFFu) << 16);        // tensor_dim0[15:0]
  g1[2] = (int)((tensor_d0 >> 16) | ((tensor_d1 & 0xFFFFu) << 16));
  g1[3] = (int)((tensor_d1 >> 16) | ((tile_d0 & 0xFFFFu) << 16));
  g1[4] = (int)(tile_d1 & 0xFFFFu);                  // tile_dim1, tile_dim2=0
  g1[5] = (int)stride_d0_elems;                      // tensor_dim0_stride[31:0]
  g1[6] = 0;
  g1[7] = 0;
  i32x4 z4 = {0, 0, 0, 0};
  i32x8 z8 = {0, 0, 0, 0, 0, 0, 0, 0};
  __builtin_amdgcn_tensor_load_to_lds(g0, g1, z4, z4, z8, 0);
}
#endif

static __device__ __forceinline__ unsigned lds_offset_of(const void* p) {
  // generic pointers to LDS carry the LDS byte offset in the low 32 bits
  return (unsigned)(uintptr_t)p;
}

// ----------------------------------------------- f32 -> f16 (transposing)
// src is K x N row-major; dst is N x K row-major (i.e. B^T), so that GEMM
// B-operand reads become contiguous 8-half (b128) runs.
__global__ void cvt_f16_T_kernel(const float* __restrict__ src,
                                 _Float16* __restrict__ dst, int Kdim, int Ndim) {
  int i = blockIdx.x * 256 + threadIdx.x;
  if (i < Kdim * Ndim) {
    int k = i / Ndim, nn = i - k * Ndim;
    dst[(size_t)nn * Kdim + k] = (_Float16)src[i];
  }
}

// ------------------------------------------------- per-node feature pipeline
__global__ __launch_bounds__(128) void node_feat_kernel(
    const float* __restrict__ x, const float* __restrict__ pos,
    const float* __restrict__ vel, const float* __restrict__ energy,
    const float* __restrict__ dW1, const float* __restrict__ db1,
    const float* __restrict__ dW2, const float* __restrict__ db2,
    const float* __restrict__ vW1, const float* __restrict__ vb1,
    const float* __restrict__ vg,  const float* __restrict__ vbeta,
    const float* __restrict__ vW2, const float* __restrict__ vb2,
    const float* __restrict__ eW1, const float* __restrict__ eb1,
    const float* __restrict__ eW2, const float* __restrict__ eb2,
    _Float16* __restrict__ feat16)
{
  __shared__ float dist[N_];
  __shared__ float red[128];
  __shared__ float hbuf[128];
  __shared__ float vin[6];

  const int b = blockIdx.x / N_;
  const int i = blockIdx.x % N_;
  const int t = threadIdx.x;
  const size_t row = (size_t)b * N_ + i;

  const float px = pos[row * 3 + 0], py = pos[row * 3 + 1], pz = pos[row * 3 + 2];

  float lsum = 0.f;
  for (int j = t; j < N_; j += 128) {
    const size_t rj = (size_t)b * N_ + j;
    float dx = px - pos[rj * 3 + 0];
    float dy = py - pos[rj * 3 + 1];
    float dz = pz - pos[rj * 3 + 2];
    float d = sqrtf(dx * dx + dy * dy + dz * dz + 1e-12f);
    dist[j] = d;
    lsum += d;
  }
  red[t] = lsum;
  __syncthreads();
  #pragma unroll
  for (int s = 64; s >= 1; s >>= 1) {
    if (t < s) red[t] += red[t + s];
    __syncthreads();
  }
  const float sumd = red[0];
  __syncthreads();

  if (t == 0) {
    float bd[4] = {3e38f, 3e38f, 3e38f, 3e38f};
    int bi[4] = {0, 0, 0, 0};
    for (int j = 0; j < N_; ++j) {
      float d = dist[j];
      if (d < bd[3]) {
        int p = 3;
        while (p > 0 && d < bd[p - 1]) { bd[p] = bd[p - 1]; bi[p] = bi[p - 1]; --p; }
        bd[p] = d; bi[p] = j;
      }
    }
    float mx = 0.f, my = 0.f, mz = 0.f;
    #pragma unroll
    for (int nn = 1; nn < 4; ++nn) {
      const size_t rn = (size_t)b * N_ + bi[nn];
      mx += vel[rn * 3 + 0]; my += vel[rn * 3 + 1]; mz += vel[rn * 3 + 2];
    }
    mx *= (1.f / 3.f); my *= (1.f / 3.f); mz *= (1.f / 3.f);
    const float vx = vel[row * 3 + 0], vy = vel[row * 3 + 1], vz = vel[row * 3 + 2];
    vin[0] = vx; vin[1] = vy; vin[2] = vz;
    vin[3] = vx - mx; vin[4] = vy - my; vin[5] = vz - mz;
  }
  __syncthreads();

  const float avg_d = sumd / (float)(N_ - 1);
  const float comm_cost = 0.1f + 0.001f * (sumd / (float)N_);

  if (t < 64) hbuf[t] = fmaxf(avg_d * dW1[t] + db1[t], 0.f);
  __syncthreads();
  float dfv = db2[t];
  for (int k = 0; k < 64; ++k) dfv += hbuf[k] * dW2[k * 128 + t];
  __syncthreads();

  const float e0 = energy[row];
  if (t < 64)
    hbuf[t] = fmaxf(e0 * eW1[t] + comm_cost * eW1[64 + t] + 0.5f * eW1[128 + t] + eb1[t], 0.f);
  __syncthreads();
  float efv = eb2[t];
  for (int k = 0; k < 64; ++k) efv += hbuf[k] * eW2[k * 128 + t];
  __syncthreads();

  float hv = vb1[t];
  #pragma unroll
  for (int c = 0; c < 6; ++c) hv += vin[c] * vW1[c * 128 + t];
  hv = fmaxf(hv, 0.f);
  red[t] = hv;
  __syncthreads();
  #pragma unroll
  for (int s = 64; s >= 1; s >>= 1) { if (t < s) red[t] += red[t + s]; __syncthreads(); }
  const float mean = red[0] * (1.f / 128.f);
  __syncthreads();
  const float dcen = hv - mean;
  red[t] = dcen * dcen;
  __syncthreads();
  #pragma unroll
  for (int s = 64; s >= 1; s >>= 1) { if (t < s) red[t] += red[t + s]; __syncthreads(); }
  const float var = red[0] * (1.f / 128.f);
  __syncthreads();
  hbuf[t] = dcen * rsqrtf(var + 1e-5f) * vg[t] + vbeta[t];
  __syncthreads();
  float vfv = vb2[t];
  for (int k = 0; k < 128; ++k) vfv += hbuf[k] * vW2[k * 128 + t];

  feat16[row * 512 + t]       = (_Float16)x[row * 128 + t];
  feat16[row * 512 + 128 + t] = (_Float16)dfv;
  feat16[row * 512 + 256 + t] = (_Float16)vfv;
  feat16[row * 512 + 384 + t] = (_Float16)efv;
}

// -------------------------------------------------------- attention gate mask
__global__ void mask_kernel(const float* __restrict__ pos,
                            unsigned char* __restrict__ mask) {
  size_t id = (size_t)blockIdx.x * 256 + threadIdx.x;
  if (id >= (size_t)B_ * N_ * N_) return;
  int j = (int)(id % N_);
  int i = (int)((id / N_) % N_);
  int b = (int)(id / ((size_t)N_ * N_));
  const size_t ri = (size_t)b * N_ + i, rj = (size_t)b * N_ + j;
  float dx = pos[ri * 3 + 0] - pos[rj * 3 + 0];
  float dy = pos[ri * 3 + 1] - pos[rj * 3 + 1];
  float dz = pos[ri * 3 + 2] - pos[rj * 3 + 2];
  float d = sqrtf(dx * dx + dy * dy + dz * dz + 1e-12f);
  mask[id] = (d <= 100.f || i == j) ? (unsigned char)1 : (unsigned char)0;
}

// ----------------------------------------------------------- LN over 256 dims
__global__ __launch_bounds__(256) void ln256_kernel(
    const float* __restrict__ h2, const float* __restrict__ g,
    const float* __restrict__ beta, _Float16* __restrict__ out) {
  __shared__ float red[256];
  const int row = blockIdx.x;
  const int t = threadIdx.x;
  const float v = h2[(size_t)row * 256 + t];
  red[t] = v;
  __syncthreads();
  #pragma unroll
  for (int s = 128; s >= 1; s >>= 1) { if (t < s) red[t] += red[t + s]; __syncthreads(); }
  const float mean = red[0] * (1.f / 256.f);
  __syncthreads();
  const float d = v - mean;
  red[t] = d * d;
  __syncthreads();
  #pragma unroll
  for (int s = 128; s >= 1; s >>= 1) { if (t < s) red[t] += red[t + s]; __syncthreads(); }
  const float var = red[0] * (1.f / 256.f);
  out[(size_t)row * 256 + t] = (_Float16)(d * rsqrtf(var + 1e-5f) * g[t] + beta[t]);
}

// ------------------------------------------------ generic WMMA GEMM (f16->f32)
// C[M x NCOLS] = A[M x K] * B[K x NCOLS] + bias, with B supplied TRANSPOSED
// (BT is NCOLS x K) so each lane's 16 B-elements are two 8-half runs, same k order as A.
// block = 256 (8 waves); wave owns a 16x16 tile of one 16-row slab (TDM-staged).
// WF16T writes the f16 output transposed (col-major) as one packed b128/lane.
template <int K, int NCOLS, bool RELU, bool WF32, bool WF16, bool WF16T>
__global__ __launch_bounds__(256) void gemm16_kernel(
    const _Float16* __restrict__ A, const _Float16* __restrict__ BT,
    const float* __restrict__ bias, float* __restrict__ Cf,
    _Float16* __restrict__ Ch)
{
  __shared__ __align__(16) _Float16 Asub[16 * K];
  const int r0 = blockIdx.x * 16;
  const int tid = threadIdx.x;

  // prefetch the BT row panel this block consumes (global_prefetch_b8)
  for (int r = tid; r < 128; r += 256)
    __builtin_prefetch(BT + (size_t)(blockIdx.y * 128 + r) * K, 0, 1);

#if HAVE_TDM
  if (tid < 32) {  // wave 0 issues a single 1-row, 8-byte-element TDM transfer
    tdm_load_2d(lds_offset_of(Asub), A + (size_t)r0 * K, /*8B*/3,
                /*tile_d0=*/4u * K, /*tile_d1=*/1u,
                /*tensor_d0=*/4u * K, /*tensor_d1=*/1u,
                /*stride=*/4u * K);
    __builtin_amdgcn_s_wait_tensorcnt(0);
  }
#else
  {
    const uint4* src = (const uint4*)(A + (size_t)r0 * K);
    uint4* dst = (uint4*)Asub;
    const int total = (16 * K) / 8;
    for (int idx = tid; idx < total; idx += 256) dst[idx] = src[idx];
  }
#endif
  __syncthreads();

  const int wid = tid >> 5;
  const int lane = tid & 31;
  const int c0 = blockIdx.y * 128 + wid * 16;
  if (c0 >= NCOLS) return;
  const int n = lane & 15;
  const bool hi = lane >= 16;
  const int ashift = hi ? 8 : 0;
  const int bshift = hi ? 8 : 0;
  const _Float16* btrow = BT + (size_t)(c0 + n) * K;

  v8f acc = {};
  #pragma unroll
  for (int kb = 0; kb < K; kb += 32) {
    v16h a = load_a_tile(Asub + (size_t)(lane & 15) * K + kb + ashift);
    v16h bmat = load_a_tile(btrow + kb + bshift);
    acc = __builtin_amdgcn_wmma_f32_16x16x32_f16(false, a, false, bmat,
                                                 (short)0, acc, false, false);
  }

  const float bv = bias[c0 + n];
  if (WF16T) {
    // transposed f16 store: 8 consecutive rows of one column -> packed b128
    union { uint4 u; _Float16 h8[8]; } t;
    #pragma unroll
    for (int e = 0; e < 8; ++e) {
      float val = acc[e] + bv;
      if (RELU) val = fmaxf(val, 0.f);
      t.h8[e] = (_Float16)val;
    }
    const int Mrows = gridDim.x * 16;
    *(uint4*)(Ch + (size_t)(c0 + n) * Mrows + r0 + (hi ? 8 : 0)) = t.u;
  }
  if (WF32 || WF16) {
    #pragma unroll
    for (int e = 0; e < 8; ++e) {
      const int rowg = r0 + (hi ? e + 8 : e);
      float val = acc[e] + bv;
      if (RELU) val = fmaxf(val, 0.f);
      const size_t off = (size_t)rowg * NCOLS + c0 + n;
      if (WF32) Cf[off] = val;
      if (WF16) Ch[off] = (_Float16)val;
    }
  }
}

// ------------------------------------------------------------------ attention
// 1 wave per (b, h, 16-query tile). Per-head K slice staged to LDS via TDM
// (strided 2D tile: 16 halves x 512 rows, stride 128 halves). QK^T via WMMA
// (DH=16 padded to K=32), masked softmax in LDS, P@V via WMMA with V supplied
// transposed (Vt is E x M) so B-operand reads are contiguous.
__global__ __launch_bounds__(32) void attn_kernel(
    const _Float16* __restrict__ Q, const _Float16* __restrict__ Kt,
    const _Float16* __restrict__ Vt, const unsigned char* __restrict__ mask,
    _Float16* __restrict__ O)
{
  __shared__ __align__(16) float    S[16][N_];     // 32 KB scores
  __shared__ __align__(16) _Float16 P[16][N_];     // 16 KB probabilities
  __shared__ __align__(16) _Float16 Ksh[N_][DH_];  // 16 KB per-head K slice
  __shared__ __align__(16) _Float16 Kz[16];        // zero row for hi-lane padding
  const int blk = blockIdx.x;
  const int qt = blk & 31;
  const int h  = (blk >> 5) & 7;
  const int b  = blk >> 8;
  const int lane = threadIdx.x;
  const int nn = lane & 15;
  const bool hi = lane >= 16;

  const _Float16* kbase = Kt + ((size_t)b * N_) * E_ + h * DH_;
#if HAVE_TDM
  tdm_load_2d(lds_offset_of(Ksh), kbase, /*2B*/1,
              /*tile_d0=*/DH_, /*tile_d1=*/N_,
              /*tensor_d0=*/DH_, /*tensor_d1=*/N_, /*stride=*/E_);
#else
  for (int r = lane; r < N_; r += 32) {
    const uint4* s = (const uint4*)(kbase + (size_t)r * E_);
    uint4* d = (uint4*)&Ksh[r][0];
    d[0] = s[0]; d[1] = s[1];
  }
#endif
  if (lane < 16) Kz[lane] = (_Float16)0.f;

  // Q tile as WMMA A operand, pre-scaled by 1/sqrt(DH)=0.25. DH=16 padded to
  // K=32: lo lanes hold K=0..15 contiguously; hi lanes (K>=16) are zero.
  v16h a;
  {
    const size_t qrow = ((size_t)b * N_ + qt * 16 + nn) * E_ + h * DH_;
    union { uint4 u[2]; v16h v; } tq;
    tq.u[0] = hi ? uint4{0, 0, 0, 0} : *(const uint4*)(Q + qrow);
    tq.u[1] = hi ? uint4{0, 0, 0, 0} : *(const uint4*)(Q + qrow + 8);
    a = tq.v * (_Float16)0.25f;
  }

#if HAVE_TDM
  __builtin_amdgcn_s_wait_tensorcnt(0);
#endif
  __syncthreads();

  for (int kt = 0; kt < N_ / 16; ++kt) {
    // B tile: lo lanes read a contiguous 16-half K row; hi lanes (K=16..31,
    // the zero padding) read the zeroed row -- address select, no divergence.
    const _Float16* krow = hi ? Kz : &Ksh[kt * 16 + nn][0];
    v16h bmat = load_row16(krow);
    v8f s = {};
    s = __builtin_amdgcn_wmma_f32_16x16x32_f16(false, a, false, bmat,
                                               (short)0, s, false, false);
    #pragma unroll
    for (int e = 0; e < 8; ++e)
      S[hi ? e + 8 : e][kt * 16 + nn] = s[e];
  }
  __syncthreads();

  for (int r = 0; r < 16; ++r) {
    const size_t mrow = ((size_t)b * N_ + qt * 16 + r) * N_;
    float mx = -3.0e38f;
    for (int c = lane; c < N_; c += 32) {
      float v = mask[mrow + c] ? S[r][c] : -1.0e9f;
      S[r][c] = v;
      mx = fmaxf(mx, v);
    }
    mx = warpMax(mx);
    float sum = 0.f;
    for (int c = lane; c < N_; c += 32) {
      float e = __expf(S[r][c] - mx);
      P[r][c] = (_Float16)e;
      sum += e;
    }
    sum = warpSum(sum);
    const float inv = 1.f / sum;
    for (int c = lane; c < N_; c += 32)
      P[r][c] = (_Float16)((float)P[r][c] * inv);
  }
  __syncthreads();

  // O = P @ V: A from P (two b128 ds loads), B from Vt (two b128 global loads)
  v8f acc = {};
  const int ashift = hi ? 8 : 0;
  const _Float16* vrow = Vt + (size_t)(h * DH_ + nn) * (B_ * N_) + (size_t)b * N_;
  for (int kt = 0; kt < N_ / 32; ++kt) {
    v16h ap = load_a_tile(&P[nn][0] + kt * 32 + ashift);
    v16h bvv = load_a_tile(vrow + kt * 32 + ashift);
    acc = __builtin_amdgcn_wmma_f32_16x16x32_f16(false, ap, false, bvv,
                                                 (short)0, acc, false, false);
  }
  #pragma unroll
  for (int e = 0; e < 8; ++e) {
    int m = hi ? e + 8 : e;
    O[((size_t)b * N_ + qt * 16 + m) * E_ + h * DH_ + nn] = (_Float16)acc[e];
  }
}

// ---------------------------------------------------------------------- launch
extern "C" void kernel_launch(void* const* d_in, const int* in_sizes, int n_in,
                              void* d_out, int out_size, void* d_ws, size_t ws_size,
                              hipStream_t stream) {
  (void)in_sizes; (void)n_in; (void)out_size; (void)ws_size;
  const float* x      = (const float*)d_in[0];
  const float* pos    = (const float*)d_in[1];
  const float* vel    = (const float*)d_in[2];
  const float* energy = (const float*)d_in[3];
  const float* dW1 = (const float*)d_in[4],  *db1 = (const float*)d_in[5];
  const float* dW2 = (const float*)d_in[6],  *db2 = (const float*)d_in[7];
  const float* vW1 = (const float*)d_in[8],  *vb1 = (const float*)d_in[9];
  const float* vg  = (const float*)d_in[10], *vbeta = (const float*)d_in[11];
  const float* vW2 = (const float*)d_in[12], *vb2 = (const float*)d_in[13];
  const float* eW1 = (const float*)d_in[14], *eb1 = (const float*)d_in[15];
  const float* eW2 = (const float*)d_in[16], *eb2 = (const float*)d_in[17];
  // d_in[18..21] = cW1,cb1,cW2,cb2: dead (only mask>0 matters) -- skipped
  const float* pW1 = (const float*)d_in[22], *pb1 = (const float*)d_in[23];
  const float* pg  = (const float*)d_in[24], *pbeta = (const float*)d_in[25];
  const float* pW2 = (const float*)d_in[26], *pb2 = (const float*)d_in[27];
  const float* Wq = (const float*)d_in[28], *bq = (const float*)d_in[29];
  const float* Wk = (const float*)d_in[30], *bk = (const float*)d_in[31];
  const float* Wv = (const float*)d_in[32], *bv = (const float*)d_in[33];
  const float* Wo = (const float*)d_in[34], *bo = (const float*)d_in[35];

  const int M = B_ * N_;
  char* ws = (char*)d_ws;
  size_t off = 0;
  auto take = [&](size_t bytes) { char* p = ws + off; off += (bytes + 255) & ~(size_t)255; return p; };
  _Float16* feat16 = (_Float16*)take((size_t)M * 512 * 2);
  unsigned char* mask8 = (unsigned char*)take((size_t)B_ * N_ * N_);
  float*    h2     = (float*)take((size_t)M * 256 * 4);
  _Float16* h2n16  = (_Float16*)take((size_t)M * 256 * 2);
  _Float16* pf16   = (_Float16*)take((size_t)M * 128 * 2);
  _Float16* q16    = (_Float16*)take((size_t)M * 128 * 2);
  _Float16* k16    = (_Float16*)take((size_t)M * 128 * 2);
  _Float16* v16t   = (_Float16*)take((size_t)M * 128 * 2);  // transposed: E x M
  _Float16* o16    = (_Float16*)take((size_t)M * 128 * 2);
  _Float16* pW1t   = (_Float16*)take((size_t)512 * 256 * 2);
  _Float16* pW2t   = (_Float16*)take((size_t)256 * 128 * 2);
  _Float16* Wqt    = (_Float16*)take((size_t)128 * 128 * 2);
  _Float16* Wkt    = (_Float16*)take((size_t)128 * 128 * 2);
  _Float16* Wvt    = (_Float16*)take((size_t)128 * 128 * 2);
  _Float16* Wot    = (_Float16*)take((size_t)128 * 128 * 2);

  cvt_f16_T_kernel<<<(512 * 256 + 255) / 256, 256, 0, stream>>>(pW1, pW1t, 512, 256);
  cvt_f16_T_kernel<<<(256 * 128 + 255) / 256, 256, 0, stream>>>(pW2, pW2t, 256, 128);
  cvt_f16_T_kernel<<<(128 * 128 + 255) / 256, 256, 0, stream>>>(Wq, Wqt, 128, 128);
  cvt_f16_T_kernel<<<(128 * 128 + 255) / 256, 256, 0, stream>>>(Wk, Wkt, 128, 128);
  cvt_f16_T_kernel<<<(128 * 128 + 255) / 256, 256, 0, stream>>>(Wv, Wvt, 128, 128);
  cvt_f16_T_kernel<<<(128 * 128 + 255) / 256, 256, 0, stream>>>(Wo, Wot, 128, 128);

  node_feat_kernel<<<B_ * N_, 128, 0, stream>>>(
      x, pos, vel, energy, dW1, db1, dW2, db2, vW1, vb1, vg, vbeta, vW2, vb2,
      eW1, eb1, eW2, eb2, feat16);

  mask_kernel<<<(int)(((size_t)B_ * N_ * N_ + 255) / 256), 256, 0, stream>>>(pos, mask8);

  gemm16_kernel<512, 256, true, true, false, false><<<dim3(M / 16, 2), 256, 0, stream>>>(
      feat16, pW1t, pb1, h2, nullptr);
  ln256_kernel<<<M, 256, 0, stream>>>(h2, pg, pbeta, h2n16);
  gemm16_kernel<256, 128, false, false, true, false><<<dim3(M / 16, 1), 256, 0, stream>>>(
      h2n16, pW2t, pb2, nullptr, pf16);
  gemm16_kernel<128, 128, false, false, true, false><<<dim3(M / 16, 1), 256, 0, stream>>>(
      pf16, Wqt, bq, nullptr, q16);
  gemm16_kernel<128, 128, false, false, true, false><<<dim3(M / 16, 1), 256, 0, stream>>>(
      pf16, Wkt, bk, nullptr, k16);
  gemm16_kernel<128, 128, false, false, false, true><<<dim3(M / 16, 1), 256, 0, stream>>>(
      pf16, Wvt, bv, nullptr, v16t);
  attn_kernel<<<B_ * H_ * (N_ / 16), 32, 0, stream>>>(q16, k16, v16t, mask8, o16);
  gemm16_kernel<128, 128, false, true, false, false><<<dim3(M / 16, 1), 256, 0, stream>>>(
      o16, Wot, bo, (float*)d_out, nullptr);
}
